// DotProductAttention_21887153340554
// MI455X (gfx1250) — compile-verified, matrix operand order checked by reading, not run on hardware
//
#include <hip/hip_runtime.h>
#include <hip/hip_bf16.h>
#include <math.h>

// ---------------------------------------------------------------------------
// Flash-attention (row-masked softmax per the reference's faithful bug) for
// MI455X / gfx1250: wave32, v_wmma_f32_16x16x32_bf16, async-DMA (ASYNCcnt)
// LDS tile staging from a pre-converted bf16 workspace.
// ---------------------------------------------------------------------------

typedef __bf16 bf16;
typedef __attribute__((ext_vector_type(16))) __bf16 v16bf;
typedef __attribute__((ext_vector_type(8)))  __bf16 bf8;
typedef __attribute__((ext_vector_type(8)))  float  v8f;
typedef __attribute__((ext_vector_type(4)))  float  f4;
typedef __attribute__((ext_vector_type(2)))  unsigned u32x2;
typedef __attribute__((ext_vector_type(4)))  int    v4i;

typedef __attribute__((address_space(1))) v4i* gptr128_t;   // global int4*
typedef __attribute__((address_space(3))) v4i* lptr128_t;   // LDS int4*

#define BATCH 16
#define SEQ   2048
#define DIM   128
#define KTILE 32                 // keys per iteration
#define WAVES 4
#define QPW   16                 // queries per wave
#define NITER (SEQ / KTILE)      // 64
#define MASKV (-1000000.0f)
#define NEGBIG (-3.0e38f)

// LDS pitches (bf16 elements): rows 16B-aligned for b128 fragment reads,
// pitches chosen so 16 consecutive rows hit distinct bank groups.
#define KT_PITCH 136             // 272B/row
#define VT_PITCH 40              // 80B/row
#define PB_PITCH 40

union frag16 { bf8 h[2]; v16bf v; };

static __device__ inline unsigned pk2bf(float a, float b) {
  union { bf16 h[2]; unsigned u; } x;
  x.h[0] = (bf16)a; x.h[1] = (bf16)b;      // -> v_cvt_pk_bf16_f32
  return x.u;
}

// 16B global -> LDS async DMA (GLOBAL_LOAD_ASYNC_TO_LDS_B128, ASYNCcnt).
// Probe-confirmed signature: (int4 AS1* src, int4 AS3* dst, int off, int cpol).
static __device__ inline void async_cp16(void* lds, const void* g) {
#if __has_builtin(__builtin_amdgcn_global_load_async_to_lds_b128)
  __builtin_amdgcn_global_load_async_to_lds_b128(
      (gptr128_t)(size_t)g, (lptr128_t)(unsigned)(size_t)lds, 0, 0);
#else
  asm volatile("global_load_async_to_lds_b128 %0, %1, off"
               :
               : "v"((unsigned)(size_t)lds), "v"((unsigned long long)(size_t)g)
               : "memory");
#endif
}

static __device__ inline void wait_async_le8() {
#if __has_builtin(__builtin_amdgcn_s_wait_asynccnt)
  __builtin_amdgcn_s_wait_asynccnt(8);
#else
  asm volatile("s_wait_asynccnt 0x8" ::: "memory");
#endif
}
static __device__ inline void wait_async_le0() {
#if __has_builtin(__builtin_amdgcn_s_wait_asynccnt)
  __builtin_amdgcn_s_wait_asynccnt(0);
#else
  asm volatile("s_wait_asynccnt 0x0" ::: "memory");
#endif
}

// --------------------------- pre-pass kernels ------------------------------

// K (f32, [b][key][d]) -> Kbf (bf16, same layout). 4 elements/thread.
__global__ __launch_bounds__(256)
void cvt_k_kernel(const float* __restrict__ K, bf16* __restrict__ Kbf) {
  const size_t quad = (size_t)blockIdx.x * blockDim.x + threadIdx.x;
  const size_t base = quad * 4;
  if (base < (size_t)BATCH * SEQ * DIM) {
    f4 x = *(const f4*)(K + base);
    u32x2 uu;
    uu.x = pk2bf(x.x, x.y);
    uu.y = pk2bf(x.z, x.w);
    *(u32x2*)(Kbf + base) = uu;
  }
}

// V (f32, [b][key][d]) -> Vtb (bf16, [b][d][key]) tiled transpose.
__global__ __launch_bounds__(256)
void cvt_vt_kernel(const float* __restrict__ V, bf16* __restrict__ Vtb) {
  __shared__ float t[32][33];
  const int tx = threadIdx.x;        // 0..31
  const int ty = threadIdx.y;        // 0..7
  const int bk = blockIdx.x;         // key tile (SEQ/32)
  const int bd = blockIdx.y;         // d tile (DIM/32)
  const int b  = blockIdx.z;
  const float* Vb = V + (size_t)b * SEQ * DIM;
  bf16* Ob = Vtb + (size_t)b * DIM * SEQ;
  #pragma unroll
  for (int r = 0; r < 4; ++r) {
    const int key = bk * 32 + ty + 8 * r;
    const int d   = bd * 32 + tx;
    t[ty + 8 * r][tx] = Vb[(size_t)key * DIM + d];   // coalesced over d
  }
  __syncthreads();
  #pragma unroll
  for (int r = 0; r < 4; ++r) {
    const int d   = bd * 32 + ty + 8 * r;
    const int key = bk * 32 + tx;
    Ob[(size_t)d * SEQ + key] = (bf16)t[tx][ty + 8 * r];  // coalesced over key
  }
}

// ------------------------------ main kernel --------------------------------

__global__ __launch_bounds__(128)
void attn_rowmask_fa_kernel(const float* __restrict__ Q,
                            const bf16*  __restrict__ Kbf,   // [b][key][d]
                            const bf16*  __restrict__ Vtb,   // [b][d][key]
                            const int*   __restrict__ VL,
                            float*       __restrict__ Out) {
  __shared__ __align__(16) bf16 Kt[2][KTILE * KT_PITCH];       // [key][d]
  __shared__ __align__(16) bf16 Vt[2][DIM * VT_PITCH];         // [d][key]
  __shared__ __align__(16) bf16 Pb[WAVES * QPW * PB_PITCH];    // per-wave P transpose

  const int tid  = threadIdx.x;
  const int lane = tid & 31;
  const int wave = tid >> 5;
  const int n16  = lane & 15;    // N index of B/C/D fragments
  const int hi   = lane >> 4;    // K-layout half select

  const int blocks_per_batch = SEQ / (WAVES * QPW);            // 32
  const int b     = blockIdx.x / blocks_per_batch;
  const int qblk  = blockIdx.x % blocks_per_batch;
  const int qbase = qblk * (WAVES * QPW) + wave * QPW;

  const int vlen = VL[b];

  const float* Qb  = Q   + (size_t)b * SEQ * DIM;
  const bf16*  Kb  = Kbf + (size_t)b * SEQ * DIM;
  const bf16*  Vtb_b = Vtb + (size_t)b * DIM * SEQ;

  // ---- Q tile -> 4 bf16 A-fragments (16x32), pre-scaled by 1/sqrt(D) ----
  const float scale = 0.08838834764831845f;   // 1/sqrt(128)
  v16bf aq[4];
  {
    const float* qrow = Qb + (size_t)(qbase + n16) * DIM;
    #pragma unroll
    for (int cd = 0; cd < 4; ++cd) {
      const int r0 = 32 * cd + 8 * hi;
      const int r1 = r0 + 16;
      f4 x0 = *(const f4*)(qrow + r0);
      f4 x1 = *(const f4*)(qrow + r0 + 4);
      f4 y0 = *(const f4*)(qrow + r1);
      f4 y1 = *(const f4*)(qrow + r1 + 4);
      frag16 u;
      u.h[0][0]=(bf16)(x0.x*scale); u.h[0][1]=(bf16)(x0.y*scale);
      u.h[0][2]=(bf16)(x0.z*scale); u.h[0][3]=(bf16)(x0.w*scale);
      u.h[0][4]=(bf16)(x1.x*scale); u.h[0][5]=(bf16)(x1.y*scale);
      u.h[0][6]=(bf16)(x1.z*scale); u.h[0][7]=(bf16)(x1.w*scale);
      u.h[1][0]=(bf16)(y0.x*scale); u.h[1][1]=(bf16)(y0.y*scale);
      u.h[1][2]=(bf16)(y0.z*scale); u.h[1][3]=(bf16)(y0.w*scale);
      u.h[1][4]=(bf16)(y1.x*scale); u.h[1][5]=(bf16)(y1.y*scale);
      u.h[1][6]=(bf16)(y1.z*scale); u.h[1][7]=(bf16)(y1.w*scale);
      aq[cd] = u.v;
    }
  }

  // ---- async DMA staging of one K/V tile (8 x b128 per thread) ----
  auto stage_async = [&](int kbase, bf16* KtP, bf16* VtP) {
    #pragma unroll
    for (int t = 0; t < 4; ++t) {            // K: 32 rows x 16 chunks of 16B
      const int idx = tid + 128 * t;         // 0..511
      const int r = idx >> 4, c = idx & 15;
      async_cp16(&KtP[r * KT_PITCH + c * 8],
                 Kb + (size_t)(kbase + r) * DIM + c * 8);
    }
    #pragma unroll
    for (int t = 0; t < 4; ++t) {            // V^T: 128 rows x 4 chunks of 16B
      const int idx = tid + 128 * t;
      const int d = idx >> 2, c = idx & 3;
      async_cp16(&VtP[d * VT_PITCH + c * 8],
                 Vtb_b + (size_t)d * SEQ + kbase + c * 8);
    }
  };

  // ---- accumulators ----
  v8f o[8];
  const v8f vzero = {0.f,0.f,0.f,0.f,0.f,0.f,0.f,0.f};
  #pragma unroll
  for (int c = 0; c < 8; ++c) o[c] = vzero;
  float m_run[8], l_run[8];
  #pragma unroll
  for (int v = 0; v < 8; ++v) { m_run[v] = NEGBIG; l_run[v] = 0.0f; }

  // ---- prologue: DMA tile 0 ----
  stage_async(0, (bf16*)Kt[0], (bf16*)Vt[0]);

  for (int kb = 0; kb < NITER; ++kb) {
    const int p = kb & 1;
    const bf16* KtP = Kt[p];
    const bf16* VtP = Vt[p];

    if (kb + 1 < NITER) {
      stage_async((kb + 1) * KTILE, (bf16*)Kt[p ^ 1], (bf16*)Vt[p ^ 1]);
      wait_async_le8();          // async loads retire in order: tile p resident
    } else {
      wait_async_le0();
    }
    __syncthreads();             // all waves' DMA portions visible

    // ---- S = Q K^T, two 16-key halves, double-buffered B fragments ----
    v8f s1 = vzero, s2 = vzero;
    frag16 fb[2][2];
    auto ldk = [&](int cd, int buf) {
      const int doff = 32 * cd + 8 * hi;
      const bf16* r0p = &KtP[n16 * KT_PITCH + doff];
      const bf16* r1p = &KtP[(16 + n16) * KT_PITCH + doff];
      fb[buf][0].h[0] = *(const bf8*)(r0p);
      fb[buf][0].h[1] = *(const bf8*)(r0p + 16);
      fb[buf][1].h[0] = *(const bf8*)(r1p);
      fb[buf][1].h[1] = *(const bf8*)(r1p + 16);
    };
    ldk(0, 0);
    #pragma unroll
    for (int cd = 0; cd < 4; ++cd) {
      const int cur = cd & 1;
      if (cd < 3) ldk(cd + 1, cur ^ 1);
      s1 = __builtin_amdgcn_wmma_f32_16x16x32_bf16(false, aq[cd], false, fb[cur][0].v,
                                                   (short)0, s1, false, false);
      s2 = __builtin_amdgcn_wmma_f32_16x16x32_bf16(false, aq[cd], false, fb[cur][1].v,
                                                   (short)0, s2, false, false);
    }

    // ---- row mask + online softmax (C layout: lane=n16, VGPR v -> row v+8hi) ----
    bf16* pb = &Pb[(wave * QPW) * PB_PITCH];
    #pragma unroll
    for (int v = 0; v < 8; ++v) {
      const int m = v + 8 * hi;
      const bool masked = (qbase + m) >= vlen;   // reference row-mask bug
      float a  = masked ? MASKV : s1[v];
      float c2 = masked ? MASKV : s2[v];
      float t = fmaxf(a, c2);
      t = fmaxf(t, __shfl_xor(t, 1, 32));
      t = fmaxf(t, __shfl_xor(t, 2, 32));
      t = fmaxf(t, __shfl_xor(t, 4, 32));
      t = fmaxf(t, __shfl_xor(t, 8, 32));
      const float mnew = fmaxf(m_run[v], t);
      const float corr = __expf(m_run[v] - mnew);
      const float p1 = __expf(a - mnew);
      const float p2 = __expf(c2 - mnew);
      float rs = p1 + p2;
      rs += __shfl_xor(rs, 1, 32);
      rs += __shfl_xor(rs, 2, 32);
      rs += __shfl_xor(rs, 4, 32);
      rs += __shfl_xor(rs, 8, 32);
      l_run[v] = l_run[v] * corr + rs;
      m_run[v] = mnew;
      #pragma unroll
      for (int c = 0; c < 8; ++c) o[c][v] *= corr;
      pb[m * PB_PITCH + n16]      = (bf16)p1;    // C-layout -> LDS (transpose)
      pb[m * PB_PITCH + 16 + n16] = (bf16)p2;
    }
    // intra-wave LDS store->load is in-order; no barrier needed for Pb

    // ---- O += P(16x32) @ V(32x128), double-buffered V fragments ----
    frag16 ap;
    {
      const bf16* prow = &Pb[(wave * QPW + n16) * PB_PITCH];
      ap.h[0] = *(const bf8*)(prow + 8 * hi);
      ap.h[1] = *(const bf8*)(prow + 16 + 8 * hi);
    }
    frag16 bv[2];
    auto ldv = [&](int c, int buf) {
      const bf16* vrow = &VtP[(16 * c + n16) * VT_PITCH];
      bv[buf].h[0] = *(const bf8*)(vrow + 8 * hi);
      bv[buf].h[1] = *(const bf8*)(vrow + 16 + 8 * hi);
    };
    ldv(0, 0);
    #pragma unroll
    for (int c = 0; c < 8; ++c) {
      const int cur = c & 1;
      if (c < 7) ldv(c + 1, cur ^ 1);
      o[c] = __builtin_amdgcn_wmma_f32_16x16x32_bf16(false, ap.v, false, bv[cur].v,
                                                     (short)0, o[c], false, false);
    }

    __syncthreads();             // tile-p reads done before DMA overwrites it
  }

  // ---- epilogue: divide by running sum, store fp32 ----
  float* obase = Out + (size_t)b * SEQ * DIM;
  #pragma unroll
  for (int v = 0; v < 8; ++v) {
    const int m = v + 8 * hi;
    const float inv = 1.0f / l_run[v];
    float* orow = obase + (size_t)(qbase + m) * DIM;
    #pragma unroll
    for (int c = 0; c < 8; ++c) {
      orow[16 * c + n16] = o[c][v] * inv;
    }
  }
}

extern "C" void kernel_launch(void* const* d_in, const int* in_sizes, int n_in,
                              void* d_out, int out_size, void* d_ws, size_t ws_size,
                              hipStream_t stream) {
  const float* q  = (const float*)d_in[0];
  const float* k  = (const float*)d_in[1];
  const float* v  = (const float*)d_in[2];
  const int*   vl = (const int*)d_in[3];
  float* out = (float*)d_out;

  // workspace: Kbf (bf16 [b][key][d]) then Vtb (bf16 [b][d][key]); 16.8MB total
  bf16* kbf = (bf16*)d_ws;
  bf16* vtb = kbf + (size_t)BATCH * SEQ * DIM;

  {   // K -> bf16
    const size_t quads = (size_t)BATCH * SEQ * DIM / 4;
    cvt_k_kernel<<<(unsigned)((quads + 255) / 256), 256, 0, stream>>>(k, kbf);
  }
  {   // V -> bf16 transposed
    dim3 g(SEQ / 32, DIM / 32, BATCH);
    dim3 t(32, 8);
    cvt_vt_kernel<<<g, t, 0, stream>>>(v, vtb);
  }

  const int blocks = BATCH * (SEQ / (WAVES * QPW));   // 512
  attn_rowmask_fa_kernel<<<blocks, WAVES * 32, 0, stream>>>(q, kbf, vtb, vl, out);
}
